// q_x_class_5772436046290
// MI455X (gfx1250) — matrix operand order repaired and checked by reading, not compiled, not run on hardware
//
#include <hip/hip_runtime.h>

#define NROWS 262144
#define YF 48

typedef __attribute__((ext_vector_type(16))) _Float16 v16h;
typedef __attribute__((ext_vector_type(8)))  float    v8f;

__device__ __forceinline__ float softplusf(float x) {
  return fmaxf(x, 0.f) + log1pf(expf(-fabsf(x)));
}

// ---------------- zero stats region ----------------
__global__ void k_zero(float* __restrict__ p, int n) {
  int i = blockIdx.x * blockDim.x + threadIdx.x;
  if (i < n) p[i] = 0.f;
}

// ---------------- one-hot -> packed index + group counts ----------------
__global__ void k_idx(const float* __restrict__ FB, const float* __restrict__ FC,
                      const float* __restrict__ RS, int* __restrict__ idx,
                      int* __restrict__ counts) {
  int n = blockIdx.x * blockDim.x + threadIdx.x;
  if (n >= NROWS) return;
  int fb = 0, fc = 0, rs = 0;
#pragma unroll
  for (int g = 1; g < 8; ++g)  if (FB[(size_t)n * 8  + g] > 0.5f) fb = g;
#pragma unroll
  for (int g = 1; g < 4; ++g)  if (FC[(size_t)n * 4  + g] > 0.5f) fc = g;
  for (int g = 1; g < 64; ++g) if (RS[(size_t)n * 64 + g] > 0.5f) rs = g;
  idx[n] = fb | (fc << 8) | (rs << 16);
  atomicAdd(&counts[fb], 1);
  atomicAdd(&counts[8 + fc], 1);
  atomicAdd(&counts[12 + rs], 1);
}

// ---------------- stage A: per-FB sums of z, w, z^2 ----------------
__global__ void k_stageA(const float* __restrict__ z, const float* __restrict__ w,
                         const int* __restrict__ idx, float* __restrict__ sumsA) {
  __shared__ float s[8 * YF * 3];
  for (int i = threadIdx.x; i < 8 * YF * 3; i += blockDim.x) s[i] = 0.f;
  __syncthreads();
  for (int n = blockIdx.x * blockDim.x + threadIdx.x; n < NROWS;
       n += gridDim.x * blockDim.x) {
    int fb = idx[n] & 255;
    const float* zr = z + (size_t)n * YF;
    const float* wr = w + (size_t)n * YF;
    float* sg = s + fb * YF * 3;
    for (int m = 0; m < YF; ++m) {
      float zv = zr[m], wv = wr[m];
      atomicAdd(&sg[m * 3 + 0], zv);
      atomicAdd(&sg[m * 3 + 1], wv);
      atomicAdd(&sg[m * 3 + 2], zv * zv);
    }
  }
  __syncthreads();
  for (int i = threadIdx.x; i < 8 * YF * 3; i += blockDim.x) atomicAdd(&sumsA[i], s[i]);
}

// paramsA[(g*48+m)*4] = {mu1z, invsd1z, mu1w, invsd1w}
__global__ void k_finA(const float* __restrict__ sumsA, const int* __restrict__ counts,
                       float* __restrict__ pA) {
  int i = threadIdx.x;  // 384 = 8*48
  if (i >= 8 * YF) return;
  int g = i / YF;
  float inv = 1.f / fmaxf((float)counts[g], 1.f);
  float Sz = sumsA[i * 3 + 0], Sw = sumsA[i * 3 + 1], Szz = sumsA[i * 3 + 2];
  float muz = Sz * inv;
  float varz = fmaxf(Szz * inv - muz * muz, 1e-20f);
  float muw = Sw * inv;
  float varw = fmaxf(Szz * inv - 2.f * muw * (Sz * inv) + muw * muw, 1e-20f);
  pA[i * 4 + 0] = muz; pA[i * 4 + 1] = rsqrtf(varz);
  pA[i * 4 + 2] = muw; pA[i * 4 + 3] = rsqrtf(varw);
}

// ---------------- stage B: per-FC sums of bz, bz^2, bw, bw^2 ----------------
__global__ void k_stageB(const float* __restrict__ z, const int* __restrict__ idx,
                         const float* __restrict__ pA, float* __restrict__ sumsB) {
  __shared__ float s[4 * YF * 4];
  __shared__ float spA[8 * YF * 4];
  for (int i = threadIdx.x; i < 4 * YF * 4; i += blockDim.x) s[i] = 0.f;
  for (int i = threadIdx.x; i < 8 * YF * 4; i += blockDim.x) spA[i] = pA[i];
  __syncthreads();
  for (int n = blockIdx.x * blockDim.x + threadIdx.x; n < NROWS;
       n += gridDim.x * blockDim.x) {
    int v = idx[n];
    int fb = v & 255, fc = (v >> 8) & 255;
    const float* zr = z + (size_t)n * YF;
    for (int m = 0; m < YF; ++m) {
      const float* pa = &spA[(fb * YF + m) * 4];
      float zv = zr[m];
      float bz = (zv - pa[0]) * pa[1];
      float bw = (zv - pa[2]) * pa[3];
      float* sg = &s[(fc * YF + m) * 4];
      atomicAdd(&sg[0], bz); atomicAdd(&sg[1], bz * bz);
      atomicAdd(&sg[2], bw); atomicAdd(&sg[3], bw * bw);
    }
  }
  __syncthreads();
  for (int i = threadIdx.x; i < 4 * YF * 4; i += blockDim.x) atomicAdd(&sumsB[i], s[i]);
}

__global__ void k_finB(const float* __restrict__ sumsB, const int* __restrict__ counts,
                       float* __restrict__ pB) {
  int i = threadIdx.x;  // 192 = 4*48
  if (i >= 4 * YF) return;
  int g = i / YF;
  float inv = 1.f / fmaxf((float)counts[8 + g], 1.f);
  float mz = sumsB[i * 4 + 0] * inv;
  float vz = fmaxf(sumsB[i * 4 + 1] * inv - mz * mz, 1e-20f);
  float mw = sumsB[i * 4 + 2] * inv;
  float vw = fmaxf(sumsB[i * 4 + 3] * inv - mw * mw, 1e-20f);
  pB[i * 4 + 0] = mz; pB[i * 4 + 1] = rsqrtf(vz);
  pB[i * 4 + 2] = mw; pB[i * 4 + 3] = rsqrtf(vw);
}

// ---------------- stage C: per-RS sums of cz, cz^2, cw, cw^2 ----------------
__global__ void k_stageC(const float* __restrict__ z, const int* __restrict__ idx,
                         const float* __restrict__ pA, const float* __restrict__ pB,
                         float* __restrict__ sumsC) {
  __shared__ float s[64 * YF * 4];     // 48 KB
  __shared__ float spA[8 * YF * 4];    // 6 KB
  __shared__ float spB[4 * YF * 4];    // 3 KB
  for (int i = threadIdx.x; i < 64 * YF * 4; i += blockDim.x) s[i] = 0.f;
  for (int i = threadIdx.x; i < 8 * YF * 4; i += blockDim.x) spA[i] = pA[i];
  for (int i = threadIdx.x; i < 4 * YF * 4; i += blockDim.x) spB[i] = pB[i];
  __syncthreads();
  for (int n = blockIdx.x * blockDim.x + threadIdx.x; n < NROWS;
       n += gridDim.x * blockDim.x) {
    int v = idx[n];
    int fb = v & 255, fc = (v >> 8) & 255, rs = (v >> 16) & 255;
    const float* zr = z + (size_t)n * YF;
    for (int m = 0; m < YF; ++m) {
      const float* pa = &spA[(fb * YF + m) * 4];
      const float* pb = &spB[(fc * YF + m) * 4];
      float zv = zr[m];
      float cz = ((zv - pa[0]) * pa[1] - pb[0]) * pb[1];
      float cw = ((zv - pa[2]) * pa[3] - pb[2]) * pb[3];
      float* sg = &s[(rs * YF + m) * 4];
      atomicAdd(&sg[0], cz); atomicAdd(&sg[1], cz * cz);
      atomicAdd(&sg[2], cw); atomicAdd(&sg[3], cw * cw);
    }
  }
  __syncthreads();
  for (int i = threadIdx.x; i < 64 * YF * 4; i += blockDim.x) atomicAdd(&sumsC[i], s[i]);
}

__global__ void k_finC(const float* __restrict__ sumsC, const int* __restrict__ counts,
                       float* __restrict__ pC) {
  for (int i = threadIdx.x; i < 64 * YF; i += blockDim.x) {
    int g = i / YF;
    float inv = 1.f / fmaxf((float)counts[12 + g], 1.f);
    float mz = sumsC[i * 4 + 0] * inv;
    float vz = fmaxf(sumsC[i * 4 + 1] * inv - mz * mz, 1e-20f);
    float mw = sumsC[i * 4 + 2] * inv;
    float vw = fmaxf(sumsC[i * 4 + 3] * inv - mw * mw, 1e-20f);
    pC[i * 4 + 0] = mz; pC[i * 4 + 1] = rsqrtf(vz);
    pC[i * 4 + 2] = mw; pC[i * 4 + 3] = rsqrtf(vw);
  }
}

// ---------------- denoised_y = (ew>0) ? softplus(ez) : 0, f16 padded [N,64] ----------------
__global__ void k_y(const float* __restrict__ z, const int* __restrict__ idx,
                    const float* __restrict__ pA, const float* __restrict__ pB,
                    const float* __restrict__ pC, _Float16* __restrict__ y16) {
  __shared__ float spA[8 * YF * 4];
  __shared__ float spB[4 * YF * 4];
  __shared__ float spC[64 * YF * 4];
  for (int i = threadIdx.x; i < 8 * YF * 4; i += blockDim.x) spA[i] = pA[i];
  for (int i = threadIdx.x; i < 4 * YF * 4; i += blockDim.x) spB[i] = pB[i];
  for (int i = threadIdx.x; i < 64 * YF * 4; i += blockDim.x) spC[i] = pC[i];
  __syncthreads();
  int n = blockIdx.x * blockDim.x + threadIdx.x;
  if (n >= NROWS) return;
  int v = idx[n];
  int fb = v & 255, fc = (v >> 8) & 255, rs = (v >> 16) & 255;
  const float* zr = z + (size_t)n * YF;
  _Float16* yr = y16 + (size_t)n * 64;
  for (int m = 0; m < YF; ++m) {
    const float* pa = &spA[(fb * YF + m) * 4];
    const float* pb = &spB[(fc * YF + m) * 4];
    const float* pc = &spC[(rs * YF + m) * 4];
    float zv = zr[m];
    float ez = (((zv - pa[0]) * pa[1] - pb[0]) * pb[1] - pc[0]) * pc[1];
    float ew = (((zv - pa[2]) * pa[3] - pb[2]) * pb[3] - pc[2]) * pc[3];
    float y = (ew > 0.f) ? softplusf(ez) : 0.f;
    yr[m] = (_Float16)y;
  }
  for (int m = YF; m < 64; ++m) yr[m] = (_Float16)0.f;
}

// ---------------- pack W (f32 [K,Ncols] row-major) into WMMA-B fragment layout ----------------
// PW[((kt*NoutPad)+n)*32 + kk] = W[(kt*32+kk)*Ncols + n], zero-padded.
__global__ void k_packw(const float* __restrict__ W, _Float16* __restrict__ PW,
                        int Kreal, int ktiles, int NoutPad, int Ncols) {
  int i = blockIdx.x * blockDim.x + threadIdx.x;
  int total = ktiles * NoutPad * 32;
  if (i >= total) return;
  int kk = i & 31;
  int rest = i >> 5;
  int n = rest % NoutPad;
  int kt = rest / NoutPad;
  int k = kt * 32 + kk;
  float v = (k < Kreal && n < Ncols) ? W[(size_t)k * Ncols + n] : 0.f;
  PW[i] = (_Float16)v;
}

// ---------------- WMMA GEMM: out = act( A[M,K] * W[K,N] + b ) ----------------
// One wave: 16 rows x (16*SUB) cols. Block 256 thr = 8 waves = 128 rows.
// grid = (NOUT/(16*SUB), M/128). All shape params compile-time: straight-line
// wmma chain, in-place accumulation, fully unrolled k-loop so the scheduler
// hoists global_load_b128 across k-steps. Large SUB shrinks grid.x so the
// (HBM-resident) activation matrix is read fewer times.
template <int KT, int NOUT, int SUB, int LDA, int LDO, int NVALID, bool FINAL>
__global__ __launch_bounds__(256) void k_gemm(
    const _Float16* __restrict__ A, const _Float16* __restrict__ PW,
    const float* __restrict__ bias, _Float16* __restrict__ O16,
    float* __restrict__ O32) {
  int lane = threadIdx.x & 31;
  int wave = threadIdx.x >> 5;
  int rowbase = blockIdx.y * 128 + wave * 16;
  int colbase = blockIdx.x * (16 * SUB);
  int arow = rowbase + (lane & 15);
  int kbaseA = (lane & 16) ? 8 : 0;    // A: lanes16-31 hold K 8..15 / 24..31
  int kbaseB = (lane & 16) ? 8 : 0;    // B: lanes16-31 hold K 8..15 / 24..31

  v8f acc[SUB] = {};
  const _Float16* abase = A + (size_t)arow * LDA + kbaseA;
  const _Float16* bbase = PW + (size_t)(colbase + (lane & 15)) * 32 + kbaseB;

#pragma unroll
  for (int kt = 0; kt < KT; ++kt) {
    v16h a;
    ((uint4*)&a)[0] = *(const uint4*)(abase + kt * 32);       // K kbaseA..+7
    ((uint4*)&a)[1] = *(const uint4*)(abase + kt * 32 + 16);  // K 16+kbaseA..+7
#pragma unroll
    for (int s = 0; s < SUB; ++s) {
      const _Float16* bp = bbase + (size_t)(kt * NOUT + 16 * s) * 32;
      v16h b;
      ((uint4*)&b)[0] = *(const uint4*)(bp);
      ((uint4*)&b)[1] = *(const uint4*)(bp + 16);
      acc[s] = __builtin_amdgcn_wmma_f32_16x16x32_f16(
          false, a, false, b, (short)0, acc[s], false, false);
    }
  }

  int mbase = rowbase + ((lane & 16) ? 8 : 0);
#pragma unroll
  for (int s = 0; s < SUB; ++s) {
    int col = colbase + 16 * s + (lane & 15);
    float bv = (!FINAL || col < NVALID) ? bias[col < NVALID ? col : 0] : 0.f;
#pragma unroll
    for (int r = 0; r < 8; ++r) {
      float v = acc[s][r] + bv;
      int row = mbase + r;
      if (!FINAL) {
        v = v > 0.f ? v : 0.2f * v;  // LeakyReLU(0.2)
        O16[(size_t)row * LDO + col] = (_Float16)v;
      } else if (col < NVALID) {
        O32[(size_t)row * NVALID + col] = v;
      }
    }
  }
}

// ---------------- host ----------------
static inline size_t al256(size_t x) { return (x + 255) & ~(size_t)255; }

extern "C" void kernel_launch(void* const* d_in, const int* in_sizes, int n_in,
                              void* d_out, int out_size, void* d_ws, size_t ws_size,
                              hipStream_t stream) {
  (void)in_sizes; (void)n_in; (void)out_size; (void)ws_size;
  const float* z  = (const float*)d_in[0];
  const float* w  = (const float*)d_in[1];
  const float* FB = (const float*)d_in[11];
  const float* FC = (const float*)d_in[12];
  const float* RS = (const float*)d_in[13];
  const float* W1 = (const float*)d_in[14]; const float* b1 = (const float*)d_in[15];
  const float* W2 = (const float*)d_in[16]; const float* b2 = (const float*)d_in[17];
  const float* W3 = (const float*)d_in[18]; const float* b3 = (const float*)d_in[19];
  const float* W4 = (const float*)d_in[20]; const float* b4 = (const float*)d_in[21];

  char* ws = (char*)d_ws;
  size_t off = 0;
  size_t o_counts = off; off = al256(off + 76 * sizeof(int));
  size_t o_sumsA  = off; off = al256(off + 8 * YF * 3 * sizeof(float));
  size_t o_sumsB  = off; off = al256(off + 4 * YF * 4 * sizeof(float));
  size_t o_sumsC  = off; off = al256(off + 64 * YF * 4 * sizeof(float));
  size_t statsEnd = off;
  size_t o_pA  = off; off = al256(off + 8 * YF * 4 * sizeof(float));
  size_t o_pB  = off; off = al256(off + 4 * YF * 4 * sizeof(float));
  size_t o_pC  = off; off = al256(off + 64 * YF * 4 * sizeof(float));
  size_t o_idx = off; off = al256(off + (size_t)NROWS * sizeof(int));
  size_t o_pw1 = off; off = al256(off + (size_t)2 * 512 * 32 * 2);
  size_t o_pw2 = off; off = al256(off + (size_t)16 * 256 * 32 * 2);
  size_t o_pw3 = off; off = al256(off + (size_t)8 * 128 * 32 * 2);
  size_t o_pw4 = off; off = al256(off + (size_t)4 * 32 * 32 * 2);
  size_t o_y16 = off; off = al256(off + (size_t)NROWS * 64 * 2);
  size_t o_bufA = off; off = al256(off + (size_t)NROWS * 512 * 2);  // h1, then h3
  size_t o_bufB = off; off = al256(off + (size_t)NROWS * 256 * 2);  // h2

  int* counts  = (int*)(ws + o_counts);
  float* sumsA = (float*)(ws + o_sumsA);
  float* sumsB = (float*)(ws + o_sumsB);
  float* sumsC = (float*)(ws + o_sumsC);
  float* pA = (float*)(ws + o_pA);
  float* pB = (float*)(ws + o_pB);
  float* pC = (float*)(ws + o_pC);
  int* idx = (int*)(ws + o_idx);
  _Float16* pw1 = (_Float16*)(ws + o_pw1);
  _Float16* pw2 = (_Float16*)(ws + o_pw2);
  _Float16* pw3 = (_Float16*)(ws + o_pw3);
  _Float16* pw4 = (_Float16*)(ws + o_pw4);
  _Float16* y16 = (_Float16*)(ws + o_y16);
  _Float16* h1  = (_Float16*)(ws + o_bufA);
  _Float16* h2  = (_Float16*)(ws + o_bufB);
  _Float16* h3  = (_Float16*)(ws + o_bufA);

  dim3 blk(256);
  int nz = (int)(statsEnd / 4);
  k_zero<<<(nz + 255) / 256, blk, 0, stream>>>((float*)ws, nz);
  k_idx<<<NROWS / 256, blk, 0, stream>>>(FB, FC, RS, idx, counts);

  k_stageA<<<512, blk, 0, stream>>>(z, w, idx, sumsA);
  k_finA<<<1, 384, 0, stream>>>(sumsA, counts, pA);
  k_stageB<<<512, blk, 0, stream>>>(z, idx, pA, sumsB);
  k_finB<<<1, 192, 0, stream>>>(sumsB, counts, pB);
  k_stageC<<<256, blk, 0, stream>>>(z, idx, pA, pB, sumsC);
  k_finC<<<1, 256, 0, stream>>>(sumsC, counts, pC);

  k_y<<<NROWS / 256, blk, 0, stream>>>(z, idx, pA, pB, pC, y16);

  k_packw<<<(2 * 512 * 32 + 255) / 256, blk, 0, stream>>>(W1, pw1, 48, 2, 512, 512);
  k_packw<<<(16 * 256 * 32 + 255) / 256, blk, 0, stream>>>(W2, pw2, 512, 16, 256, 256);
  k_packw<<<(8 * 128 * 32 + 255) / 256, blk, 0, stream>>>(W3, pw3, 256, 8, 128, 128);
  k_packw<<<(4 * 32 * 32 + 255) / 256, blk, 0, stream>>>(W4, pw4, 128, 4, 32, 24);

  // Layer 1: [N,64(pad48)] x [64,512]; wave tile 16x128 -> grid.x=4
  k_gemm<2, 512, 8, 64, 512, 512, false>
      <<<dim3(4, NROWS / 128), blk, 0, stream>>>(y16, pw1, b1, h1, nullptr);
  // Layer 2: [N,512] x [512,256]; wave tile 16x256 -> grid.x=1, h1 read once
  k_gemm<16, 256, 16, 512, 256, 256, false>
      <<<dim3(1, NROWS / 128), blk, 0, stream>>>(h1, pw2, b2, h2, nullptr);
  // Layer 3: [N,256] x [256,128]; wave tile 16x128 -> grid.x=1, h2 read once
  k_gemm<8, 128, 8, 256, 128, 128, false>
      <<<dim3(1, NROWS / 128), blk, 0, stream>>>(h2, pw3, b3, h3, nullptr);
  // Layer 4 (final, f32 out, Nout 24 padded to 32): [N,128] x [128,32]
  k_gemm<4, 32, 2, 128, 24, 24, true>
      <<<dim3(1, NROWS / 128), blk, 0, stream>>>(h3, pw4, b4, nullptr, (float*)d_out);
}
